// CrossAttentionBlock_26096221290813
// MI455X (gfx1250) — compile-verified, matrix operand order checked by reading, not run on hardware
//
#include <hip/hip_runtime.h>

// ---------------------------------------------------------------------------
// CrossAttentionBlock for MI455X (gfx1250, wave32, WMMA).
// Pipeline (all on `stream`):
//   1) Kbf  = bf16(context @ Wk + bk)            [B*M, H]  row-major
//   2) Vt   = bf16(context @ Wv + bv)^T          [B, H, M] (transposed store)
//   3) Qbf  = bf16(x @ Wq + bq)                  [B*N, H]  row-major
//   4) Ctx  = flash-attention(Qbf, Kbf, Vt)      [B*N, H]  bf16
//   5) out  = f32(Ctx @ Wo + bo) + x             [B*N, H]  f32
// All matmuls: v_wmma_f32_16x16x32_bf16 (f32 accumulate).
// ---------------------------------------------------------------------------

#define HID   512
#define CTXD  768
#define BB    8
#define NN    4096
#define MM    1024

typedef __attribute__((ext_vector_type(16))) __bf16 bf16x16;
typedef __attribute__((ext_vector_type(8)))  __bf16 bf16x8;
typedef __attribute__((ext_vector_type(8)))  float  f32x8;

// ---- CDNA5 async global->LDS copy (ASYNCcnt-tracked), guarded -------------
// Toolchain declares: void(v4i addrspace(1)*, v4i addrspace(3)*, Imm, Imm)
#if __has_builtin(__builtin_amdgcn_global_load_async_to_lds_b128)
#define HAVE_ASYNC_LDS 1
typedef int v4i_vs __attribute__((__vector_size__(16)));
__device__ __forceinline__ void async_copy_b128(const void* g, void* l) {
  __builtin_amdgcn_global_load_async_to_lds_b128(
      (__attribute__((address_space(1))) v4i_vs*)g,
      (__attribute__((address_space(3))) v4i_vs*)l, 0, 0);
}
#endif

__device__ __forceinline__ void wait_async_zero() {
#if __has_builtin(__builtin_amdgcn_s_wait_asynccnt)
  __builtin_amdgcn_s_wait_asynccnt(0);
#else
  asm volatile("s_wait_asynccnt 0x0" ::: "memory");
#endif
}

__device__ __forceinline__ f32x8 zero8() {
  f32x8 z = {0.f, 0.f, 0.f, 0.f, 0.f, 0.f, 0.f, 0.f};
  return z;
}

__device__ __forceinline__ f32x8 wmma_bf16(bf16x16 a, bf16x16 b, f32x8 c) {
  // (neg_a, A, neg_b, B, c_mod, C, reuse_a, reuse_b)
  return __builtin_amdgcn_wmma_f32_16x16x32_bf16(false, a, false, b, (short)0,
                                                 c, false, false);
}

// Load one 16-lane-group WMMA operand fragment (16x32 A or 32x16 B) from a
// row-major bf16 array. ISA layout: lanes 0-15 hold rows 0-15 with K half-pairs
// {0..7,16..23}; lanes 16-31 hold rows 0-15 with K {8..15,24..31}.
// => per lane: two contiguous 16-byte chunks at k0+8*lanegrp and +16.
__device__ __forceinline__ bf16x16 load_frag(const __bf16* base, int stride,
                                             int k0, int lane) {
  const int lg = (lane >> 4) & 1;
  const int ln = lane & 15;
  const __bf16* p = base + (size_t)ln * stride + k0 + lg * 8;
  bf16x8 lo = *(const bf16x8*)(p);
  bf16x8 hi = *(const bf16x8*)(p + 16);
  bf16x16 f;
#pragma unroll
  for (int i = 0; i < 8; ++i) { f[i] = lo[i]; f[i + 8] = hi[i]; }
  return f;
}

// ---------------------------------------------------------------------------
// Generic projection GEMM:  out = A[R,Kdim] @ W[Kdim,Hout] + bias  (+ extras)
//   A_BF16: A is bf16 (else f32, converted to bf16 on LDS stage-in)
//   SMODE : 0 = store bf16 row-major
//           1 = store bf16 transposed per batch: out[b][col][m], ldT = M
//           2 = store f32 row-major with residual add
// Block: 256 threads (8 waves), tile 64 rows x 128 cols; wave = 32x32.
// ---------------------------------------------------------------------------
template <bool A_BF16, int SMODE>
__global__ __launch_bounds__(256) void proj_kernel(
    const void* __restrict__ Aptr, const float* __restrict__ W,
    const float* __restrict__ bias, const float* __restrict__ resid,
    void* __restrict__ outp, int Kdim, int Hout, int ldT) {
  constexpr int ASTR = 72;  // padded LDS strides: 144B rows, conflict-free b128
  constexpr int WSTR = 72;
  __shared__ __bf16 sA[64 * ASTR];   // A tile  [64 rows][64 k] padded
  __shared__ __bf16 sW[128 * WSTR];  // W tile, transposed: [128 cols][64 k]

  const int tid  = threadIdx.x;
  const int wave = tid >> 5;
  const int lane = tid & 31;
  const int wr   = wave >> 2;  // 0..1
  const int wc   = wave & 3;   // 0..3
  const int r0   = blockIdx.x * 64;
  const int c0   = blockIdx.y * 128;

  const float*  Af = (const float*)Aptr;
  const __bf16* Ab = (const __bf16*)Aptr;

  f32x8 acc[2][2];
  acc[0][0] = zero8(); acc[0][1] = zero8();
  acc[1][0] = zero8(); acc[1][1] = zero8();

  for (int k0 = 0; k0 < Kdim; k0 += 64) {
    // Stage A tile (f32 or bf16 -> bf16), coalesced on col.
    for (int i = tid; i < 64 * 64; i += 256) {
      const int row = i >> 6, col = i & 63;
      const size_t g = (size_t)(r0 + row) * Kdim + k0 + col;
      const float v = A_BF16 ? (float)Ab[g] : Af[g];
      sA[row * ASTR + col] = (__bf16)v;
    }
    // Stage W tile transposed (read coalesced on Hout, scatter to LDS).
    for (int i = tid; i < 128 * 64; i += 256) {
      const int cc = i & 127, kk = i >> 7;
      sW[cc * WSTR + kk] = (__bf16)W[(size_t)(k0 + kk) * Hout + c0 + cc];
    }
    __syncthreads();
#pragma unroll
    for (int kb = 0; kb < 64; kb += 32) {
      bf16x16 a0 = load_frag(sA + (wr * 32 + 0)  * ASTR, ASTR, kb, lane);
      bf16x16 a1 = load_frag(sA + (wr * 32 + 16) * ASTR, ASTR, kb, lane);
      bf16x16 b0 = load_frag(sW + (wc * 32 + 0)  * WSTR, WSTR, kb, lane);
      bf16x16 b1 = load_frag(sW + (wc * 32 + 16) * WSTR, WSTR, kb, lane);
      acc[0][0] = wmma_bf16(a0, b0, acc[0][0]);
      acc[0][1] = wmma_bf16(a0, b1, acc[0][1]);
      acc[1][0] = wmma_bf16(a1, b0, acc[1][0]);
      acc[1][1] = wmma_bf16(a1, b1, acc[1][1]);
    }
    __syncthreads();
  }

  // Epilogue. C/D layout: element r -> row = r + 8*(lane>=16), col = lane%16.
  const int lg = lane >> 4, ln = lane & 15;
#pragma unroll
  for (int mi = 0; mi < 2; ++mi) {
#pragma unroll
    for (int ni = 0; ni < 2; ++ni) {
      const int col = c0 + wc * 32 + ni * 16 + ln;
      const float bv = bias[col];
#pragma unroll
      for (int r = 0; r < 8; ++r) {
        const int row = r0 + wr * 32 + mi * 16 + r + 8 * lg;
        const float v = acc[mi][ni][r] + bv;
        if (SMODE == 0) {
          ((__bf16*)outp)[(size_t)row * Hout + col] = (__bf16)v;
        } else if (SMODE == 1) {
          const int bb = row / ldT, m = row % ldT;
          ((__bf16*)outp)[((size_t)bb * Hout + col) * ldT + m] = (__bf16)v;
        } else {
          const size_t o = (size_t)row * Hout + col;
          ((float*)outp)[o] = v + resid[o];
        }
      }
    }
  }
}

// ---------------------------------------------------------------------------
// Flash attention: one block = 16 query rows of one batch, 8 waves.
// Per 128-wide M tile:
//   wave w computes S[:, w*16..] = Qtile @ K^T  (16 WMMA, K from global,
//     B-fragment = contiguous b128 loads from row-major K)
//   threads 0..15: online softmax (running max m, sum l, rescale factor)
//   wave w accumulates O[:, w*64..] += P @ V    (16 WMMA, B from Vt[B,H,M])
// ---------------------------------------------------------------------------
__global__ __launch_bounds__(256) void attn_kernel(
    const __bf16* __restrict__ Q, const __bf16* __restrict__ Kb,
    const __bf16* __restrict__ Vt, __bf16* __restrict__ Ctx) {
  constexpr int QSTR = 520;  // padded strides (16B-aligned, spread banks)
  constexpr int PSTR = 136;
  __shared__ __bf16 sQ[16 * QSTR];
  __shared__ float  sS[16 * 128];
  __shared__ __bf16 sP[16 * PSTR];
  __shared__ float  sM[16], sL[16], sFac[16];

  const int tid  = threadIdx.x;
  const int wave = tid >> 5;
  const int lane = tid & 31;
  const int lg   = lane >> 4;
  const int ln   = lane & 15;
  const int b    = blockIdx.x >> 8;          // N/16 == 256 tiles per batch
  const int n0   = (blockIdx.x & 255) << 4;

  const __bf16* Qbase = Q  + ((size_t)b * NN + n0) * HID;
  const __bf16* Kbase = Kb + (size_t)b * MM * HID;
  const __bf16* Vbase = Vt + (size_t)b * HID * MM;

  // Stage Q tile [16, 512] into LDS.
#ifdef HAVE_ASYNC_LDS
  // CDNA5 async DMA path: global -> LDS without VGPR round-trip (ASYNCcnt).
  for (int i = tid * 8; i < 16 * HID; i += 256 * 8) {
    const int row = i >> 9, col = i & (HID - 1);
    async_copy_b128(Qbase + i, sQ + row * QSTR + col);
  }
  wait_async_zero();
#else
  for (int i = tid * 8; i < 16 * HID; i += 256 * 8) {
    const int row = i >> 9, col = i & (HID - 1);
    *(bf16x8*)(sQ + row * QSTR + col) = *(const bf16x8*)(Qbase + i);
  }
#endif
  if (tid < 16) { sM[tid] = -1e30f; sL[tid] = 0.f; }

  f32x8 O[4];
#pragma unroll
  for (int j = 0; j < 4; ++j) O[j] = zero8();
  __syncthreads();

  const float scale = 0.04419417382415922f;  // 1/sqrt(512)

  for (int m0 = 0; m0 < MM; m0 += 128) {
    const __bf16* krow = Kbase + (size_t)(m0 + wave * 16) * HID;
    // Prefetch next M tile's K/V streams into L2 (global_prefetch_b8).
    if (m0 + 128 < MM) {
      __builtin_prefetch(krow + (size_t)128 * HID, 0, 1);
      __builtin_prefetch(Vbase + (size_t)(wave * 64) * MM + m0 + 128, 0, 1);
    }
    // ---- S slice: columns m0 + wave*16 .. +15 -----------------------------
    f32x8 s = zero8();
#pragma unroll
    for (int k0 = 0; k0 < HID; k0 += 32) {
      bf16x16 a  = load_frag(sQ, QSTR, k0, lane);
      bf16x16 bf = load_frag(krow, HID, k0, lane);  // B(k=h, n=m) = K[m][h]
      s = wmma_bf16(a, bf, s);
    }
#pragma unroll
    for (int r = 0; r < 8; ++r)
      sS[(r + 8 * lg) * 128 + wave * 16 + ln] = s[r] * scale;
    __syncthreads();

    // ---- Online softmax update (one thread per query row) -----------------
    if (tid < 16) {
      float* rowp = sS + tid * 128;
      float mx = sM[tid];
      for (int c = 0; c < 128; ++c) mx = fmaxf(mx, rowp[c]);
      const float fac = __expf(sM[tid] - mx);
      float l = sL[tid] * fac;
      for (int c = 0; c < 128; ++c) {
        const float e = __expf(rowp[c] - mx);
        l += e;
        sP[tid * PSTR + c] = (__bf16)e;
      }
      sM[tid] = mx; sL[tid] = l; sFac[tid] = fac;
    }
    __syncthreads();

    // ---- Rescale O and accumulate O += P @ V ------------------------------
    float fr[8];
#pragma unroll
    for (int r = 0; r < 8; ++r) fr[r] = sFac[r + 8 * lg];
#pragma unroll
    for (int j = 0; j < 4; ++j)
#pragma unroll
      for (int r = 0; r < 8; ++r) O[j][r] *= fr[r];

#pragma unroll
    for (int kb = 0; kb < 128; kb += 32) {
      bf16x16 aP = load_frag(sP, PSTR, kb, lane);
#pragma unroll
      for (int j = 0; j < 4; ++j) {
        // B(k=m, n=h) = Vt[h][m]; rows of Vt are contiguous in m.
        bf16x16 bV = load_frag(Vbase + (size_t)(wave * 64 + j * 16) * MM + m0,
                               MM, kb, lane);
        O[j] = wmma_bf16(aP, bV, O[j]);
      }
    }
    __syncthreads();
  }

  // ---- Normalize and store ctx_out bf16 -----------------------------------
  float inv[8];
#pragma unroll
  for (int r = 0; r < 8; ++r) inv[r] = 1.0f / sL[r + 8 * lg];
#pragma unroll
  for (int j = 0; j < 4; ++j)
#pragma unroll
    for (int r = 0; r < 8; ++r) {
      const size_t row = (size_t)b * NN + n0 + r + 8 * lg;
      const int col = wave * 64 + j * 16 + ln;
      Ctx[row * HID + col] = (__bf16)(O[j][r] * inv[r]);
    }
}

// ---------------------------------------------------------------------------
extern "C" void kernel_launch(void* const* d_in, const int* in_sizes, int n_in,
                              void* d_out, int out_size, void* d_ws,
                              size_t ws_size, hipStream_t stream) {
  (void)in_sizes; (void)n_in; (void)out_size; (void)ws_size;
  const float* x    = (const float*)d_in[0];  // [B,N,H]
  const float* ctxt = (const float*)d_in[1];  // [B,M,CTX]
  const float* Wq   = (const float*)d_in[2];
  const float* bq   = (const float*)d_in[3];
  const float* Wk   = (const float*)d_in[4];
  const float* bk   = (const float*)d_in[5];
  const float* Wv   = (const float*)d_in[6];
  const float* bv   = (const float*)d_in[7];
  const float* Wo   = (const float*)d_in[8];
  const float* bo   = (const float*)d_in[9];

  char* ws = (char*)d_ws;
  const size_t szQ = (size_t)BB * NN * HID * 2;  // 32 MB
  const size_t szK = (size_t)BB * MM * HID * 2;  //  8 MB
  __bf16* Qbf  = (__bf16*)(ws);
  __bf16* Kbf  = (__bf16*)(ws + szQ);
  __bf16* Vtb  = (__bf16*)(ws + szQ + szK);
  __bf16* Ctxb = (__bf16*)(ws + szQ + 2 * szK);

  const dim3 blk(256);
  const int RQ = BB * NN;  // 32768
  const int RK = BB * MM;  // 8192

  // K = bf16(context @ Wk + bk), row-major [B*M, H]
  proj_kernel<false, 0><<<dim3(RK / 64, HID / 128), blk, 0, stream>>>(
      ctxt, Wk, bk, nullptr, Kbf, CTXD, HID, 0);
  // V^T = bf16(context @ Wv + bv) stored [B, H, M]
  proj_kernel<false, 1><<<dim3(RK / 64, HID / 128), blk, 0, stream>>>(
      ctxt, Wv, bv, nullptr, Vtb, CTXD, HID, MM);
  // Q = bf16(x @ Wq + bq), row-major [B*N, H]
  proj_kernel<false, 0><<<dim3(RQ / 64, HID / 128), blk, 0, stream>>>(
      x, Wq, bq, nullptr, Qbf, HID, HID, 0);
  // Flash attention -> Ctx bf16 [B*N, H]
  attn_kernel<<<dim3(BB * (NN / 16)), blk, 0, stream>>>(Qbf, Kbf, Vtb, Ctxb);
  // out = f32(Ctx @ Wo + bo) + x
  proj_kernel<true, 2><<<dim3(RQ / 64, HID / 128), blk, 0, stream>>>(
      Ctxb, Wo, bo, x, d_out, HID, HID, 0);
}
